// DeformableConv2d_37572373905599
// MI455X (gfx1250) — matrix operand order repaired and checked by reading, not compiled, not run on hardware
//
#include <hip/hip_runtime.h>

typedef __attribute__((ext_vector_type(16))) _Float16 v16h;
typedef __attribute__((ext_vector_type(8)))  float    v8f;

#define HW   36864      // 192*192
#define IMG  192
#define NB   2
#define NC   64

__device__ __forceinline__ v8f wmma16(v16h a, v16h b, v8f c) {
  // v_wmma_f32_16x16x32_f16 : D = A(16x32) x B(32x16) + C, f32 accum
  return __builtin_amdgcn_wmma_f32_16x16x32_f16(false, a, false, b, (short)0, c, false, false);
}

// A-fragment: lane<16 -> row m=lane, K 0..15 (K0=0)
//             lane>=16 -> row m=lane-16, K 16..31 (K0=8). Two 16B LDS loads.
__device__ __forceinline__ v16h ldsFragA(const _Float16* row, int kbase, int K0) {
  union { uint4 q[2]; v16h h; } u;
  u.q[0] = *reinterpret_cast<const uint4*>(row + kbase + 2 * K0);
  u.q[1] = *reinterpret_cast<const uint4*>(row + kbase + 2 * K0 + 8);
  return u.h;
}

// fast sigmoid: v_exp_f32 + v_rcp_f32 (no IEEE divide expansion)
__device__ __forceinline__ float sigm(float v) {
  return __builtin_amdgcn_rcpf(1.0f + __expf(-v));
}

// Unconditional 16-channel gather: one base pointer, immediate channel offsets,
// zero-padding applied AFTER the loads (loads batch into a clause).
__device__ __forceinline__ v16h gather16(const float* basePtr, float zmask) {
  float tmp[16];
#pragma unroll
  for (int e = 0; e < 16; ++e) tmp[e] = basePtr[e * HW];
  v16h bf;
#pragma unroll
  for (int e = 0; e < 16; ++e) bf[e] = (_Float16)(tmp[e] * zmask);
  return bf;
}

// ---------------- Kernel A: grouped offset conv + 100*sigmoid-50 ----------------
// per (b,g): M=18(pad32) x K=288 x N=36864.  K order = kk*32 + cin (cin: 16 warp + 16 src)
__global__ __launch_bounds__(256) void offset_conv_kernel(
    const float* __restrict__ warp, const float* __restrict__ src,
    const float* __restrict__ offw, const float* __restrict__ offb,
    float* __restrict__ offOut)
{
  __shared__ __align__(16) _Float16 wLds[32 * 288];
  const int tid = threadIdx.x;
  for (int idx = tid; idx < 32 * 288; idx += 256) {
    int co = idx / 288, kidx = idx - co * 288;
    int kk = kidx >> 5, cin = kidx & 31;
    float v = (co < 18) ? offw[(co * 32 + cin) * 9 + kk] : 0.0f;
    wLds[idx] = (_Float16)v;
  }
  __syncthreads();

  const int blk = blockIdx.x;              // ((b*4+g)*288 + tileBlk)
  const int tileBlk = blk % 288, bg = blk / 288;
  const int g = bg & 3, b = bg >> 2;
  const int wave = tid >> 5, lane = tid & 31;
  const int half = lane >> 4, nl = lane & 15;
  const int n  = tileBlk * 128 + wave * 16 + nl;
  const int py = n / IMG, px = n - py * IMG;

  const float* chBase = (half ? src : warp) + (size_t)(b * NC + g * 16) * HW;

  v8f acc0 = {}, acc1 = {};
  const int K0 = half * 8;
  for (int s = 0; s < 9; ++s) {
    const int ki = s / 3, kj = s - 3 * (s / 3);
    const int yy = py + ki - 1, xx = px + kj - 1;
    const bool inb = (yy >= 0) & (yy < IMG) & (xx >= 0) & (xx < IMG);
    const int sp = inb ? (yy * IMG + xx) : 0;          // always-legal address
    const float zmask = inb ? 1.0f : 0.0f;             // zero-pad after load
    v16h bf = gather16(chBase + sp, zmask);
    v16h a0 = ldsFragA(&wLds[nl * 288],        s * 32, K0);
    v16h a1 = ldsFragA(&wLds[(16 + nl) * 288], s * 32, K0);
    acc0 = wmma16(a0, bf, acc0);
    acc1 = wmma16(a1, bf, acc1);
  }
  float* outB = offOut + (size_t)(b * 72 + g * 18) * HW + n;
#pragma unroll
  for (int r = 0; r < 8; ++r) {
    int co = r + 8 * half;                      // 0..15, always valid
    outB[co * HW] = 100.0f * sigm(acc0[r] + offb[co]) - 50.0f;
    int co1 = 16 + r + 8 * half;                // only 16,17 valid
    if (co1 < 18)
      outB[co1 * HW] = 100.0f * sigm(acc1[r] + offb[co1]) - 50.0f;
  }
}

// ---------------- Kernel B: modulator conv + 2*sigmoid ----------------
// M=36(pad48) x K=576 x N=36864.  K order = kk*64 + cin.
__global__ __launch_bounds__(256) void mod_conv_kernel(
    const float* __restrict__ warp, const float* __restrict__ modw,
    const float* __restrict__ modb, float* __restrict__ modOut)
{
  __shared__ __align__(16) _Float16 wLds[48 * 576];
  const int tid = threadIdx.x;
  for (int idx = tid; idx < 48 * 576; idx += 256) {
    int co = idx / 576, kidx = idx - co * 576;
    int kk = kidx >> 6, cin = kidx & 63;
    float v = (co < 36) ? modw[(co * 64 + cin) * 9 + kk] : 0.0f;
    wLds[idx] = (_Float16)v;
  }
  __syncthreads();

  const int blk = blockIdx.x;              // b*288 + tileBlk
  const int tileBlk = blk % 288, b = blk / 288;
  const int wave = tid >> 5, lane = tid & 31;
  const int half = lane >> 4, nl = lane & 15;
  const int n  = tileBlk * 128 + wave * 16 + nl;
  const int py = n / IMG, px = n - py * IMG;

  const float* base = warp + (size_t)b * NC * HW;
  v8f acc[3] = {};
  const int K0 = half * 8;
  for (int s = 0; s < 18; ++s) {
    const int kk = s >> 1;
    const int ki = kk / 3, kj = kk - 3 * (kk / 3);
    const int cinBase = (s & 1) * 32 + 16 * half;
    const int yy = py + ki - 1, xx = px + kj - 1;
    const bool inb = (yy >= 0) & (yy < IMG) & (xx >= 0) & (xx < IMG);
    const int sp = inb ? (yy * IMG + xx) : 0;
    const float zmask = inb ? 1.0f : 0.0f;
    v16h bf = gather16(base + cinBase * HW + sp, zmask);
#pragma unroll
    for (int t = 0; t < 3; ++t) {
      v16h a = ldsFragA(&wLds[(t * 16 + nl) * 576], s * 32, K0);
      acc[t] = wmma16(a, bf, acc[t]);
    }
  }
#pragma unroll
  for (int t = 0; t < 3; ++t)
#pragma unroll
    for (int r = 0; r < 8; ++r) {
      int co = t * 16 + r + 8 * half;
      if (co < 36)
        modOut[(size_t)(b * 36 + co) * HW + n] = 2.0f * sigm(acc[t][r] + modb[co]);
    }
}

// ---------------- Kernel C: modulated deformable conv ----------------
// M=64 x K=576 x N=36864. K order = (g*9+kk)*16 + c. Weights staged in 2 LDS phases (36KB).
__device__ __forceinline__ void stage_regw(_Float16* lds, const float* __restrict__ regw,
                                           int ph, int tid) {
  for (int idx = tid; idx < 64 * 288; idx += 256) {
    int o = idx / 288, kidx = idx - o * 288;
    int gk = (ph * 288 + kidx) >> 4;     // sample index p = g*9+kk (0..35)
    int c  = kidx & 15;
    int g  = gk / 9, kk = gk - g * 9;
    lds[idx] = (_Float16)regw[(o * 64 + g * 16 + c) * 9 + kk];
  }
}

__global__ __launch_bounds__(256) void deform_conv_kernel(
    const float* __restrict__ warp, const float* __restrict__ regw,
    const float* __restrict__ offMap, const float* __restrict__ modOut,
    float* __restrict__ xOut)
{
  __shared__ __align__(16) _Float16 wLds[64 * 288];
  const int tid = threadIdx.x;
  const int blk = blockIdx.x;              // b*288 + tileBlk
  const int tileBlk = blk % 288, b = blk / 288;
  const int wave = tid >> 5, lane = tid & 31;
  const int half = lane >> 4, nl = lane & 15;
  const int n  = tileBlk * 128 + wave * 16 + nl;
  const int py = n / IMG, px = n - py * IMG;

  stage_regw(wLds, regw, 0, tid);
  __syncthreads();

  v8f acc[4] = {};
  const int K0 = half * 8;
  for (int s = 0; s < 18; ++s) {
    if (s == 9) {                          // swap to K = 288..575 weights
      __syncthreads();
      stage_regw(wLds, regw, 1, tid);
      __syncthreads();
    }
    const int p  = 2 * s + half;           // sample index g*9+kk for this lane
    const int g  = p / 9, kk = p - g * 9;
    const int ki = kk / 3, kj = kk - 3 * (kk / 3);

    const float* ob = offMap + (size_t)(b * 72 + g * 18 + kk * 2) * HW + n;
    const float oy  = ob[0];
    const float ox  = ob[HW];
    const float msk = modOut[(size_t)(b * 36 + p) * HW + n];

    const float fy = oy + (float)(ki + py - 1);
    const float fx = ox + (float)(kj + px - 1);
    const float y0f = floorf(fy), x0f = floorf(fx);
    const float ly = fy - y0f, lx = fx - x0f;
    const int y0 = (int)y0f, x0 = (int)x0f;
    const int y1 = y0 + 1,   x1 = x0 + 1;
    const float vy0 = (y0 >= 0 && y0 < IMG) ? 1.0f : 0.0f;
    const float vy1 = (y1 >= 0 && y1 < IMG) ? 1.0f : 0.0f;
    const float vx0 = (x0 >= 0 && x0 < IMG) ? 1.0f : 0.0f;
    const float vx1 = (x1 >= 0 && x1 < IMG) ? 1.0f : 0.0f;
    const int cy0 = min(max(y0, 0), IMG - 1), cy1 = min(max(y1, 0), IMG - 1);
    const int cx0 = min(max(x0, 0), IMG - 1), cx1 = min(max(x1, 0), IMG - 1);
    // fold validity + mask into bilinear weights (clamped index is always legal)
    const float w00 = (1.0f - ly) * (1.0f - lx) * vy0 * vx0 * msk;
    const float w01 = (1.0f - ly) * lx          * vy0 * vx1 * msk;
    const float w10 = ly          * (1.0f - lx) * vy1 * vx0 * msk;
    const float w11 = ly          * lx          * vy1 * vx1 * msk;
    const int i00 = cy0 * IMG + cx0, i01 = cy0 * IMG + cx1;
    const int i10 = cy1 * IMG + cx0, i11 = cy1 * IMG + cx1;

    const float* cb = warp + (size_t)(b * NC + g * 16) * HW;
    // unconditional 4-corner gathers; batch loads first, then FMA chain
    float t00[16], t01[16], t10[16], t11[16];
#pragma unroll
    for (int e = 0; e < 16; ++e) t00[e] = cb[e * HW + i00];
#pragma unroll
    for (int e = 0; e < 16; ++e) t01[e] = cb[e * HW + i01];
#pragma unroll
    for (int e = 0; e < 16; ++e) t10[e] = cb[e * HW + i10];
#pragma unroll
    for (int e = 0; e < 16; ++e) t11[e] = cb[e * HW + i11];
    v16h bf;
#pragma unroll
    for (int e = 0; e < 16; ++e) {
      float v = w00 * t00[e] + w01 * t01[e] + w10 * t10[e] + w11 * t11[e];
      bf[e] = (_Float16)v;
    }
    const int sl = (s < 9) ? s : (s - 9);
#pragma unroll
    for (int t = 0; t < 4; ++t) {
      v16h a = ldsFragA(&wLds[(t * 16 + nl) * 288], sl * 32, K0);
      acc[t] = wmma16(a, bf, acc[t]);
    }
  }
#pragma unroll
  for (int t = 0; t < 4; ++t)
#pragma unroll
    for (int r = 0; r < 8; ++r) {
      int co = t * 16 + r + 8 * half;
      xOut[(size_t)(b * NC + co) * HW + n] = acc[t][r];
    }
}

extern "C" void kernel_launch(void* const* d_in, const int* in_sizes, int n_in,
                              void* d_out, int out_size, void* d_ws, size_t ws_size,
                              hipStream_t stream) {
  (void)in_sizes; (void)n_in; (void)out_size; (void)ws_size;
  const float* warp = (const float*)d_in[0];
  const float* src  = (const float*)d_in[1];
  const float* offw = (const float*)d_in[2];
  const float* offb = (const float*)d_in[3];
  const float* modw = (const float*)d_in[4];
  const float* modb = (const float*)d_in[5];
  const float* regw = (const float*)d_in[6];

  float* out    = (float*)d_out;
  float* xOut   = out;                                   // [2,64,192,192]
  float* offOut = out + (size_t)NB * NC * HW;            // [2,72,192,192]
  float* modScr = (float*)d_ws;                          // [2,36,192,192] f32

  offset_conv_kernel<<<NB * 4 * 288, 256, 0, stream>>>(warp, src, offw, offb, offOut);
  mod_conv_kernel   <<<NB * 288,     256, 0, stream>>>(warp, modw, modb, modScr);
  deform_conv_kernel<<<NB * 288,     256, 0, stream>>>(warp, regw, offOut, modScr, xOut);
}
